// IBRNetWithNeuRay_46334107189519
// MI455X (gfx1250) — matrix operand order repaired and checked by reading, not compiled, not run on hardware
//
#include <hip/hip_runtime.h>
#include <hip/hip_bf16.h>

// ---------------- problem constants ----------------
#define R_    1024
#define S_    64
#define V_    8
#define F_    35
#define NR_   32
#define RS_   (R_*S_)        // 65536
#define RSV_  (RS_*V_)       // 524288
#define KPAD  224            // 207 padded to 7 K-tiles of 32

typedef __attribute__((ext_vector_type(16))) _Float16 v16h;
typedef __attribute__((ext_vector_type(8)))  float    v8f;
typedef __attribute__((ext_vector_type(4)))  float    v4f;

union F8 { v8f v; float f[8]; };
union AV { v16h v; v4f q[2]; };

__device__ __forceinline__ float eluf(float x)  { return x > 0.f ? x : (expf(x) - 1.f); }
__device__ __forceinline__ float sigmf_(float x){ return 1.f / (1.f + expf(-x)); }

// =====================================================================
// K1: per-(r,s): similarity/total, direction MLP, neuray gate, weights,
//     fused mean/var -> write padded f16 GEMM input row (224 halves)
// one wave (32 lanes) per (r,s); 8 (r,s) per 256-thread block
// =====================================================================
__global__ __launch_bounds__(256) void k1_stage1(
    const float* __restrict__ rgb, const float* __restrict__ neuray,
    const float* __restrict__ rdiff, const float* __restrict__ mask,
    const float* __restrict__ rdw1, const float* __restrict__ rdb1,
    const float* __restrict__ rdw2, const float* __restrict__ rdb2,
    const float* __restrict__ nrw1, const float* __restrict__ nrb1,
    const float* __restrict__ nrw2, const float* __restrict__ nrb2,
    _Float16* __restrict__ xin, float* __restrict__ wgt, float* __restrict__ tot)
{
    __shared__ float s_rdw1[64], s_rdb1[16], s_rdw2[560], s_rdb2[35];
    __shared__ float s_nrw1[256], s_nrb1[8], s_nrw2[8], s_nrb2[1];
    __shared__ float s_rgb[8][8][36];
    __shared__ float s_feat[8][8][36];
    __shared__ float s_w0[8][8], s_wm[8][8];
    __shared__ float s_gf[8][140];

    int tid = threadIdx.x, wave = tid >> 5, lane = tid & 31;
    int rs = blockIdx.x * 8 + wave;

    for (int i = tid; i < 64;  i += 256) s_rdw1[i] = rdw1[i];
    for (int i = tid; i < 16;  i += 256) s_rdb1[i] = rdb1[i];
    for (int i = tid; i < 560; i += 256) s_rdw2[i] = rdw2[i];
    for (int i = tid; i < 35;  i += 256) s_rdb2[i] = rdb2[i];
    for (int i = tid; i < 256; i += 256) s_nrw1[i] = nrw1[i];
    if (tid < 8)               s_nrb1[tid]     = nrb1[tid];
    if (tid >= 8 && tid < 16)  s_nrw2[tid - 8] = nrw2[tid - 8];
    if (tid == 16)             s_nrb2[0]       = nrb2[0];

    for (int i = lane; i < 280; i += 32) {
        int v = i / 35, f = i - v * 35;
        s_rgb[wave][v][f] = rgb[(size_t)rs * 280 + i];
    }
    __syncthreads();

    // similarity -> total = tanh(sum exp(off-diagonal dots))
    float esum = 0.f;
    for (int p = lane; p < 64; p += 32) {
        int v = p >> 3, w = p & 7;
        if (v != w) {
            float d = 0.f;
            for (int f = 0; f < 35; ++f) d += s_rgb[wave][v][f] * s_rgb[wave][w][f];
            esum += expf(d);
        }
    }
    for (int o = 16; o >= 1; o >>= 1) esum += __shfl_xor(esum, o);
    if (lane == 0) tot[rs] = tanhf(esum);

    // mask weights
    float maskv = (lane < 8) ? mask[rs * 8 + lane] : 0.f;
    float msum = maskv;
    for (int o = 16; o >= 1; o >>= 1) msum += __shfl_xor(msum, o);
    float wv = maskv / (msum + 1e-8f);

    if (lane < 8) {
        int v = lane;
        // direction MLP: 4 -> 16 -> 35, ELU after each
        float rdv[4];
        for (int i = 0; i < 4; ++i) rdv[i] = rdiff[(size_t)rs * 32 + v * 4 + i];
        float h[16];
        for (int j = 0; j < 16; ++j) {
            float a = s_rdb1[j];
            for (int i = 0; i < 4; ++i) a += rdv[i] * s_rdw1[i * 16 + j];
            h[j] = eluf(a);
        }
        for (int f = 0; f < 35; ++f) {
            float a = s_rdb2[f];
            for (int j = 0; j < 16; ++j) a += h[j] * s_rdw2[j * 35 + f];
            s_feat[wave][v][f] = s_rgb[wave][v][f] + eluf(a);
        }
        // neuray gate: 32 -> 8 (ELU) -> 1 ; sigmoid * weight
        const float* nrp = neuray + (size_t)rs * 256 + v * 32;
        float h2[8];
        for (int j = 0; j < 8; ++j) {
            float a = s_nrb1[j];
            for (int i = 0; i < 32; ++i) a += nrp[i] * s_nrw1[i * 8 + j];
            h2[j] = eluf(a);
        }
        float nv = s_nrb2[0];
        for (int j = 0; j < 8; ++j) nv += h2[j] * s_nrw2[j];
        s_w0[wave][v] = sigmf_(nv) * wv;
        s_wm[wave][v] = wv;
        wgt[rs * 8 + v] = wv;
    }
    __syncthreads();

    // fused mean / variance over V, two weight sets
    for (int f = lane; f < 35; f += 32) {
        float m0 = 0.f, m1 = 0.f;
        for (int v = 0; v < 8; ++v) {
            float x = s_feat[wave][v][f];
            m0 += s_w0[wave][v] * x; m1 += s_wm[wave][v] * x;
        }
        float va0 = 0.f, va1 = 0.f;
        for (int v = 0; v < 8; ++v) {
            float x = s_feat[wave][v][f];
            float d0 = x - m0, d1 = x - m1;
            va0 += s_w0[wave][v] * d0 * d0;
            va1 += s_wm[wave][v] * d1 * d1;
        }
        s_gf[wave][f] = m0; s_gf[wave][35 + f] = va0;
        s_gf[wave][70 + f] = m1; s_gf[wave][105 + f] = va1;
    }
    __syncthreads();

    // write padded f16 GEMM-A rows: [gf(140) | feat(35) | neuray(32) | 0(17)]
    for (int i = lane; i < 8 * KPAD; i += 32) {
        int v = i / KPAD, k = i - v * KPAD;
        float val;
        if (k < 140)       val = s_gf[wave][k];
        else if (k < 175)  val = s_feat[wave][v][k - 140];
        else if (k < 207)  val = neuray[(size_t)rs * 256 + v * 32 + (k - 175)];
        else               val = 0.f;
        xin[(size_t)(rs * 8 + v) * KPAD + k] = (_Float16)val;
    }
}

// =====================================================================
// K2: WMMA GEMM  (RSV x 224) @ (224 x 64) + bias -> ELU -> f16
// 8 waves / block, each wave a 16x64 tile, K-loop 7 x 32
// =====================================================================
__global__ __launch_bounds__(256) void k2_gemm1(
    const _Float16* __restrict__ xin, const float* __restrict__ w1,
    const float* __restrict__ b1, _Float16* __restrict__ h1)
{
    __shared__ __attribute__((aligned(32))) _Float16 wt[64 * KPAD]; // transposed, zero-padded
    __shared__ float bs[64];
    int tid = threadIdx.x;
    for (int i = tid; i < 64 * KPAD; i += 256) {
        int n = i / KPAD, k = i - n * KPAD;
        wt[i] = (k < 207) ? (_Float16)w1[k * 64 + n] : (_Float16)0.f;
    }
    if (tid < 64) bs[tid] = b1[tid];
    __syncthreads();

    int wave = tid >> 5, lane = tid & 31;
    int m0 = (blockIdx.x * 8 + wave) * 16;
    int mn = lane & 15;   // A-row in tile / B,D column in tile
    int hi = lane >> 4;   // half-select per ISA f16 layouts
    const _Float16* arow = xin + (size_t)(m0 + mn) * KPAD;

    F8 acc[4] = {};
    for (int kt = 0; kt < 7; ++kt) {
        int k0 = kt * 32;
        AV a;
        a.q[0] = *(const v4f*)(arow + k0 + hi * 16);       // K = k0+16hi .. +7
        a.q[1] = *(const v4f*)(arow + k0 + hi * 16 + 8);   // K = k0+16hi+8 ..
#pragma unroll
        for (int nt = 0; nt < 4; ++nt) {
            v16h bv = *(const v16h*)&wt[(nt * 16 + mn) * KPAD + k0 + hi * 16];
            acc[nt].v = __builtin_amdgcn_wmma_f32_16x16x32_f16(
                false, a.v, false, bv, (short)0, acc[nt].v, false, false);
        }
    }
#pragma unroll
    for (int nt = 0; nt < 4; ++nt)
        for (int e = 0; e < 8; ++e) {
            int row = m0 + e + hi * 8;
            int col = nt * 16 + mn;
            float v = acc[nt].f[e] + bs[col];
            v = v > 0.f ? v : (expf(v) - 1.f);
            h1[(size_t)row * 64 + col] = (_Float16)v;
        }
}

// =====================================================================
// K3: WMMA GEMM  (RSV x 64) @ (64 x 32) + bias -> ELU -> f32 "x"
// =====================================================================
__global__ __launch_bounds__(256) void k3_gemm2(
    const _Float16* __restrict__ h1, const float* __restrict__ w2,
    const float* __restrict__ b2, float* __restrict__ xb)
{
    __shared__ __attribute__((aligned(32))) _Float16 wt[32 * 64];
    __shared__ float bs[32];
    int tid = threadIdx.x;
    for (int i = tid; i < 32 * 64; i += 256) {
        int n = i >> 6, k = i & 63;
        wt[i] = (_Float16)w2[k * 32 + n];
    }
    if (tid < 32) bs[tid] = b2[tid];
    __syncthreads();

    int wave = tid >> 5, lane = tid & 31;
    int m0 = (blockIdx.x * 8 + wave) * 16;
    int mn = lane & 15, hi = lane >> 4;
    const _Float16* arow = h1 + (size_t)(m0 + mn) * 64;

    F8 acc[2] = {};
    for (int kt = 0; kt < 2; ++kt) {
        int k0 = kt * 32;
        AV a;
        a.q[0] = *(const v4f*)(arow + k0 + hi * 16);
        a.q[1] = *(const v4f*)(arow + k0 + hi * 16 + 8);
#pragma unroll
        for (int nt = 0; nt < 2; ++nt) {
            v16h bv = *(const v16h*)&wt[(nt * 16 + mn) * 64 + k0 + hi * 16];
            acc[nt].v = __builtin_amdgcn_wmma_f32_16x16x32_f16(
                false, a.v, false, bv, (short)0, acc[nt].v, false, false);
        }
    }
    for (int nt = 0; nt < 2; ++nt)
        for (int e = 0; e < 8; ++e) {
            int row = m0 + e + hi * 8, col = nt * 16 + mn;
            float v = acc[nt].f[e] + bs[col];
            xb[(size_t)row * 32 + col] = v > 0.f ? v : (expf(v) - 1.f);
        }
}

// =====================================================================
// K4: vis / v2 / r MLP chain + reductions over V, blended RGB, gf65
// one wave per (r,s); lanes 0..7 own views; shuffle reductions
// =====================================================================
__global__ __launch_bounds__(256) void k4_stage3(
    const float* __restrict__ xb, const float* __restrict__ wgt,
    const float* __restrict__ mask, const float* __restrict__ rdiff,
    const float* __restrict__ rgb,
    const float* vw1, const float* vb1, const float* vw2, const float* vb2,
    const float* v2w1, const float* v2b1, const float* v2w2, const float* v2b2,
    const float* rw1, const float* rb1, const float* rw2, const float* rb2,
    const float* rw3, const float* rb3,
    float* __restrict__ gf65, float* __restrict__ nval, float* __restrict__ dout)
{
    __shared__ float s_vw1[1024], s_vb1[32], s_vw2[1056], s_vb2[33];
    __shared__ float s_v2w1[1024], s_v2b1[32], s_v2w2[32], s_v2b2[1];
    __shared__ float s_rw1[592], s_rb1[16], s_rw2[128], s_rb2[8], s_rw3[8], s_rb3[1];
    __shared__ float s_x[8][8][32];
    __shared__ float s_w2[8][8];

    int tid = threadIdx.x, wave = tid >> 5, lane = tid & 31;
    for (int i = tid; i < 1024; i += 256) s_vw1[i] = vw1[i];
    for (int i = tid; i < 1056; i += 256) s_vw2[i] = vw2[i];
    for (int i = tid; i < 1024; i += 256) s_v2w1[i] = v2w1[i];
    for (int i = tid; i < 592;  i += 256) s_rw1[i] = rw1[i];
    for (int i = tid; i < 128;  i += 256) s_rw2[i] = rw2[i];
    if (tid < 32) s_vb1[tid] = vb1[tid];
    if (tid < 33) s_vb2[tid] = vb2[tid];
    if (tid < 32) s_v2b1[tid] = v2b1[tid];
    if (tid < 32) s_v2w2[tid] = v2w2[tid];
    if (tid == 0) s_v2b2[0] = v2b2[0];
    if (tid < 16) s_rb1[tid] = rb1[tid];
    if (tid < 8)  s_rb2[tid] = rb2[tid];
    if (tid < 8)  s_rw3[tid] = rw3[tid];
    if (tid == 1) s_rb3[0] = rb3[0];

    int rs = blockIdx.x * 8 + wave;
    for (int i = lane; i < 256; i += 32) {
        int v = i >> 5, d = i & 31;
        s_x[wave][v][d] = xb[(size_t)(rs * 8 + v) * 32 + d];
    }
    __syncthreads();

    float maskv = (lane < 8) ? mask[rs * 8 + lane] : 0.f;
    float nv = maskv;
    for (int o = 16; o >= 1; o >>= 1) nv += __shfl_xor(nv, o);
    if (lane == 0) nval[rs] = nv;

    if (lane < 8) {
        int v = lane;
        float wv = wgt[rs * 8 + v];
        float xi[32];
        for (int d = 0; d < 32; ++d) xi[d] = s_x[wave][v][d];

        // vis MLP: (x*weight) 32 -> 32 (ELU) -> 33 (ELU)
        float h[32];
        for (int j = 0; j < 32; ++j) {
            float a = s_vb1[j];
            for (int d = 0; d < 32; ++d) a += xi[d] * wv * s_vw1[d * 32 + j];
            h[j] = eluf(a);
        }
        float xn[32];
        for (int c = 0; c < 32; ++c) {
            float a = s_vb2[c];
            for (int j = 0; j < 32; ++j) a += h[j] * s_vw2[j * 33 + c];
            xn[c] = xi[c] + eluf(a);   // x = x + x_res
        }
        float a32 = s_vb2[32];
        for (int j = 0; j < 32; ++j) a32 += h[j] * s_vw2[j * 33 + 32];
        float vis0 = sigmf_(eluf(a32)) * maskv;

        // v2: (x*vis) 32 -> 32 (ELU) -> 1, sigmoid * mask
        float h2[32];
        for (int j = 0; j < 32; ++j) {
            float a = s_v2b1[j];
            for (int d = 0; d < 32; ++d) a += xn[d] * vis0 * s_v2w1[d * 32 + j];
            h2[j] = eluf(a);
        }
        float o2 = s_v2b2[0];
        for (int j = 0; j < 32; ++j) o2 += h2[j] * s_v2w2[j];
        float vis2 = sigmf_(o2) * maskv;

        for (int d = 0; d < 32; ++d) s_x[wave][v][d] = xn[d]; // updated x

        // r MLP on [x(32), vis, ray_diff(4)] : 37 -> 16 (ELU) -> 8 (ELU) -> 1
        float rh[16];
        for (int j = 0; j < 16; ++j) {
            float a = s_rb1[j];
            for (int i2 = 0; i2 < 32; ++i2) a += xn[i2] * s_rw1[i2 * 16 + j];
            a += vis2 * s_rw1[32 * 16 + j];
            for (int i2 = 0; i2 < 4; ++i2)
                a += rdiff[(size_t)rs * 32 + v * 4 + i2] * s_rw1[(33 + i2) * 16 + j];
            rh[j] = eluf(a);
        }
        float rh2[8];
        for (int c = 0; c < 8; ++c) {
            float a = s_rb2[c];
            for (int j = 0; j < 16; ++j) a += rh[j] * s_rw2[j * 8 + c];
            rh2[c] = eluf(a);
        }
        float xro = s_rb3[0];
        for (int c = 0; c < 8; ++c) xro += rh2[c] * s_rw3[c];
        if (maskv == 0.f) xro = -1e9f;

        // weight2 = vis2 / (sum_v vis2 + 1e-8)
        float ss = vis2;
        for (int o = 4; o >= 1; o >>= 1) ss += __shfl_xor(ss, o);
        s_w2[wave][v] = vis2 / (ss + 1e-8f);

        // blending softmax over V and rgb output
        float mx = xro;
        for (int o = 4; o >= 1; o >>= 1) mx = fmaxf(mx, __shfl_xor(mx, o));
        float e = expf(xro - mx);
        float se = e;
        for (int o = 4; o >= 1; o >>= 1) se += __shfl_xor(se, o);
        float bl = e / se;
        for (int c = 0; c < 3; ++c) {
            float pc = bl * rgb[(size_t)(rs * 8 + v) * 35 + c];
            for (int o = 4; o >= 1; o >>= 1) pc += __shfl_xor(pc, o);
            if (v == 0) dout[(size_t)rs * 4 + c] = pc;
        }
    }
    __syncthreads();

    // mean/var over V of updated x with weight2 -> gf65
    {
        int d = lane;
        float m = 0.f;
        for (int v = 0; v < 8; ++v) m += s_w2[wave][v] * s_x[wave][v][d];
        float va = 0.f;
        for (int v = 0; v < 8; ++v) {
            float df = s_x[wave][v][d] - m;
            va += s_w2[wave][v] * df * df;
        }
        gf65[(size_t)rs * 65 + d] = m;
        gf65[(size_t)rs * 65 + 32 + d] = va;
        if (lane == 0) {
            float mw = 0.f;
            for (int v = 0; v < 8; ++v) mw += s_w2[wave][v];
            gf65[(size_t)rs * 65 + 64] = mw * 0.125f;
        }
    }
}

// =====================================================================
// K5: g-MLP -> gt_ibr, prompt mix, posenc, 4-head attention + LN, sigma
// one block (64 threads = 64 tokens) per ray r
// =====================================================================
__global__ __launch_bounds__(64) void k5_stage4(
    const float* __restrict__ gf65, const float* __restrict__ tot,
    const float* __restrict__ nval, const float* __restrict__ prompt,
    const float* gw1, const float* gb1, const float* gw2, const float* gb2,
    const float* wq, const float* wk, const float* wvv, const float* wfc,
    const float* lng, const float* lnb,
    const float* ogw1, const float* ogb1, const float* ogw2, const float* ogb2,
    float* __restrict__ dout)
{
    __shared__ float s_gw1[4160], s_gb1[64], s_gw2[1024], s_gb2[16];
    __shared__ float s_wq[256], s_wk[256], s_wv[256], s_wfc[256];
    __shared__ float s_lng[16], s_lnb[16], s_ogw1[256], s_ogb1[16], s_ogw2[16], s_ogb2[1];
    __shared__ float s_tok[64][16], s_k[64][16], s_v[64][16];

    int tid = threadIdx.x;
    for (int i = tid; i < 4160; i += 64) s_gw1[i] = gw1[i];
    for (int i = tid; i < 1024; i += 64) s_gw2[i] = gw2[i];
    for (int i = tid; i < 256; i += 64) {
        s_wq[i] = wq[i]; s_wk[i] = wk[i]; s_wv[i] = wvv[i];
        s_wfc[i] = wfc[i]; s_ogw1[i] = ogw1[i];
    }
    if (tid < 64) s_gb1[tid] = gb1[tid];
    if (tid < 16) {
        s_gb2[tid] = gb2[tid]; s_lng[tid] = lng[tid]; s_lnb[tid] = lnb[tid];
        s_ogb1[tid] = ogb1[tid]; s_ogw2[tid] = ogw2[tid];
    }
    if (tid == 0) s_ogb2[0] = ogb2[0];
    __syncthreads();

    int s = tid;
    int rs = blockIdx.x * 64 + s;
    const float* gin = gf65 + (size_t)rs * 65;

    // g-MLP: 65 -> 64 (ELU) -> 16 (ELU)
    float h[64];
    for (int j = 0; j < 64; ++j) {
        float a = s_gb1[j];
        for (int i = 0; i < 65; ++i) a += gin[i] * s_gw1[i * 64 + j];
        h[j] = eluf(a);
    }
    float g16[16];
    for (int c = 0; c < 16; ++c) {
        float a = s_gb2[c];
        for (int j = 0; j < 64; ++j) a += h[j] * s_gw2[j * 16 + c];
        g16[c] = eluf(a);
    }
    // gt_ibr output (before prompt mixing)
    for (int c = 0; c < 16; ++c)
        dout[(size_t)RS_ * 4 + (size_t)rs * 16 + c] = g16[c];

    float t = tot[rs];
    for (int c = 0; c < 16; ++c)
        g16[c] = t * g16[c] + (1.f - t) * prompt[(size_t)rs * 16 + c];

    // positional encoding (d_hid=16, pos=s)
    for (int j = 0; j < 16; ++j) {
        int pair = j >> 1;
        float ang = (float)s * expf(-1.15129254650564f * (float)pair); // ln(1e4)/8
        g16[j] += (j & 1) ? cosf(ang) : sinf(ang);
    }
    for (int c = 0; c < 16; ++c) s_tok[s][c] = g16[c];
    __syncthreads();

    // q,k,v projections (no bias)
    float q[16];
    for (int j = 0; j < 16; ++j) {
        float aq = 0.f, ak = 0.f, av = 0.f;
        for (int c = 0; c < 16; ++c) {
            float x = s_tok[s][c];
            aq += x * s_wq[c * 16 + j];
            ak += x * s_wk[c * 16 + j];
            av += x * s_wv[c * 16 + j];
        }
        q[j] = aq; s_k[s][j] = ak; s_v[s][j] = av;
    }
    __syncthreads();

    float am = (nval[rs] > 1.f) ? 1.f : 0.f;  // per-query mask row
    float aout[16];
    for (int hh = 0; hh < 4; ++hh) {
        float mx = -1e30f;
        for (int key = 0; key < 64; ++key) {
            float sc;
            if (am > 0.f) {
                sc = 0.f;
                for (int d = 0; d < 4; ++d) sc += q[hh * 4 + d] * s_k[key][hh * 4 + d];
                sc *= 0.5f;                  // 1/sqrt(d_k), d_k=4
            } else sc = -1e9f;
            mx = fmaxf(mx, sc);
        }
        float se = 0.f, num[4] = {0.f, 0.f, 0.f, 0.f};
        for (int key = 0; key < 64; ++key) {
            float sc;
            if (am > 0.f) {
                sc = 0.f;
                for (int d = 0; d < 4; ++d) sc += q[hh * 4 + d] * s_k[key][hh * 4 + d];
                sc *= 0.5f;
            } else sc = -1e9f;
            float e = expf(sc - mx);
            se += e;
            for (int d = 0; d < 4; ++d) num[d] += e * s_v[key][hh * 4 + d];
        }
        for (int d = 0; d < 4; ++d) aout[hh * 4 + d] = num[d] / se;
    }

    // fc + residual + layernorm
    float o[16];
    for (int c = 0; c < 16; ++c) {
        float a = 0.f;
        for (int j = 0; j < 16; ++j) a += aout[j] * s_wfc[j * 16 + c];
        o[c] = a + s_tok[s][c];
    }
    float mu = 0.f;
    for (int c = 0; c < 16; ++c) mu += o[c];
    mu *= (1.f / 16.f);
    float va = 0.f;
    for (int c = 0; c < 16; ++c) { float d = o[c] - mu; va += d * d; }
    va *= (1.f / 16.f);
    float inv = rsqrtf(va + 1e-6f);
    float y[16];
    for (int c = 0; c < 16; ++c) y[c] = (o[c] - mu) * inv * s_lng[c] + s_lnb[c];

    // sigma head: 16 -> 16 (ELU) -> 1, relu, masks
    float oh[16];
    for (int j = 0; j < 16; ++j) {
        float a = s_ogb1[j];
        for (int c = 0; c < 16; ++c) a += y[c] * s_ogw1[c * 16 + j];
        oh[j] = eluf(a);
    }
    float sg = s_ogb2[0];
    for (int j = 0; j < 16; ++j) sg += oh[j] * s_ogw2[j];
    sg = fmaxf(sg, 0.f);
    if (nval[rs] < 1.f) sg = 0.f;
    if (t < 0.2f)       sg = 0.f;
    dout[(size_t)rs * 4 + 3] = sg;
}

// =====================================================================
// host launcher
// =====================================================================
extern "C" void kernel_launch(void* const* d_in, const int* in_sizes, int n_in,
                              void* d_out, int out_size, void* d_ws, size_t ws_size,
                              hipStream_t stream) {
    (void)in_sizes; (void)n_in; (void)out_size; (void)ws_size;
    const float* rgb     = (const float*)d_in[0];
    const float* neuray  = (const float*)d_in[1];
    const float* rdiff   = (const float*)d_in[2];
    const float* mask    = (const float*)d_in[3];
    const float* prompt  = (const float*)d_in[4];
    /* d_in[5] = mean_hit_prob (unused by reference forward) */
    const float* rd_w1 = (const float*)d_in[6];  const float* rd_b1 = (const float*)d_in[7];
    const float* rd_w2 = (const float*)d_in[8];  const float* rd_b2 = (const float*)d_in[9];
    const float* b_w1  = (const float*)d_in[10]; const float* b_b1  = (const float*)d_in[11];
    const float* b_w2  = (const float*)d_in[12]; const float* b_b2  = (const float*)d_in[13];
    const float* v_w1  = (const float*)d_in[14]; const float* v_b1  = (const float*)d_in[15];
    const float* v_w2  = (const float*)d_in[16]; const float* v_b2  = (const float*)d_in[17];
    const float* v2_w1 = (const float*)d_in[18]; const float* v2_b1 = (const float*)d_in[19];
    const float* v2_w2 = (const float*)d_in[20]; const float* v2_b2 = (const float*)d_in[21];
    const float* g_w1  = (const float*)d_in[22]; const float* g_b1  = (const float*)d_in[23];
    const float* g_w2  = (const float*)d_in[24]; const float* g_b2  = (const float*)d_in[25];
    const float* og_w1 = (const float*)d_in[26]; const float* og_b1 = (const float*)d_in[27];
    const float* og_w2 = (const float*)d_in[28]; const float* og_b2 = (const float*)d_in[29];
    const float* r_w1  = (const float*)d_in[30]; const float* r_b1  = (const float*)d_in[31];
    const float* r_w2  = (const float*)d_in[32]; const float* r_b2  = (const float*)d_in[33];
    const float* r_w3  = (const float*)d_in[34]; const float* r_b3  = (const float*)d_in[35];
    const float* nr_w1 = (const float*)d_in[36]; const float* nr_b1 = (const float*)d_in[37];
    const float* nr_w2 = (const float*)d_in[38]; const float* nr_b2 = (const float*)d_in[39];
    const float* a_wq  = (const float*)d_in[40]; const float* a_wk  = (const float*)d_in[41];
    const float* a_wv  = (const float*)d_in[42]; const float* a_fc  = (const float*)d_in[43];
    const float* a_lng = (const float*)d_in[44]; const float* a_lnb = (const float*)d_in[45];

    // workspace layout (all offsets 256B-aligned)
    char* ws = (char*)d_ws;
    _Float16* xin  = (_Float16*)(ws);                         // RSV*224 f16 = 234,881,024 B
    float*    wgt  = (float*)   (ws + 234881024);             // RSV f32    =   2,097,152 B
    float*    tot  = (float*)   (ws + 236978176);             // RS f32     =     262,144 B
    float*    nval = (float*)   (ws + 237240320);             // RS f32     =     262,144 B
    _Float16* h1   = (_Float16*)(ws + 237502464);             // RSV*64 f16 =  67,108,864 B
    float*    xb   = (float*)   (ws + 304611328);             // RSV*32 f32 =  67,108,864 B
    float*    gf65 = (float*)   (ws + 371720192);             // RS*65 f32  =  17,039,360 B
    float*    dout = (float*)d_out;

    k1_stage1<<<RS_ / 8, 256, 0, stream>>>(
        rgb, neuray, rdiff, mask,
        rd_w1, rd_b1, rd_w2, rd_b2, nr_w1, nr_b1, nr_w2, nr_b2,
        xin, wgt, tot);

    k2_gemm1<<<RSV_ / 128, 256, 0, stream>>>(xin, b_w1, b_b1, h1);
    k3_gemm2<<<RSV_ / 128, 256, 0, stream>>>(h1, b_w2, b_b2, xb);

    k4_stage3<<<RS_ / 8, 256, 0, stream>>>(
        xb, wgt, mask, rdiff, rgb,
        v_w1, v_b1, v_w2, v_b2, v2_w1, v2_b1, v2_w2, v2_b2,
        r_w1, r_b1, r_w2, r_b2, r_w3, r_b3,
        gf65, nval, dout);

    k5_stage4<<<R_, 64, 0, stream>>>(
        gf65, tot, nval, prompt,
        g_w1, g_b1, g_w2, g_b2,
        a_wq, a_wk, a_wv, a_fc, a_lng, a_lnb,
        og_w1, og_b1, og_w2, og_b2,
        dout);
}
